// CAKT_13829794693844
// MI455X (gfx1250) — compile-verified, matrix operand order checked by reading, not run on hardware
//
#include <hip/hip_runtime.h>
#include <hip/hip_bf16.h>
#include <math.h>

// ---------------- problem constants (from reference) ----------------
#define BSZ   8
#define SEQ   1024
#define DMOD  256
#define NHEAD 8
#define DKH   32
#define DFFN  1024
#define NLAY  3
#define MROWS (BSZ*SEQ)   // 8192

typedef __attribute__((ext_vector_type(16))) __bf16 v16bf;
typedef __attribute__((ext_vector_type(8)))  __bf16 v8bf;
typedef __attribute__((ext_vector_type(8)))  float  v8f;

__device__ __forceinline__ v8f wmma_bf16(v16bf a, v16bf b, v8f c) {
  return __builtin_amdgcn_wmma_f32_16x16x32_bf16(false, a, false, b, (short)0, c,
                                                 false, false);
}

// K-pair base for 16-bit A/B fragments (ISA 7.12.2)
__device__ __forceinline__ int kpair(int vv, int g) {
  return ((vv & 4) << 2) | ((vv & 3) << 1) | (g << 3);
}

// A-fragment (16x32 bf16, row-major source): per lane = two contiguous 16B chunks
// chunk0: K in [8g, 8g+8), chunk1: K in [16+8g, 24+8g)   -> exactly frag elems 0..7, 8..15
template <typename PT>
__device__ __forceinline__ v16bf load_a_frag(const PT* rowp, int kc, int g) {
  v8bf lo = *(const v8bf*)(rowp + kc + 8 * g);
  v8bf hi = *(const v8bf*)(rowp + kc + 16 + 8 * g);
  return __builtin_shufflevector(lo, hi, 0, 1, 2, 3, 4, 5, 6, 7,
                                 8, 9, 10, 11, 12, 13, 14, 15);
}

// ---------------- f32 -> bf16 (plain) ----------------
__global__ void cvt_bf16(const float* __restrict__ in, __bf16* __restrict__ out, long n) {
  long i = (long)blockIdx.x * blockDim.x + threadIdx.x;
  long stride = (long)gridDim.x * blockDim.x;
  for (; i < n; i += stride) out[i] = (__bf16)in[i];
}

// ------- f32 weights -> bf16 packed in WMMA B-fragment order -------
// dest flat (per layer): ((nt*K32 + c)*32 + lane)*16 + e
//   lane -> (hx,g) col = nt*16+hx ; e -> (vv,lo) k = c*32 + kpair(vv,g) + lo
__global__ void cvt_pack_w(const float* __restrict__ W, __bf16* __restrict__ out,
                           int K, int N, long total) {
  long matsz = (long)K * N;
  int K32 = K >> 5;
  long i = (long)blockIdx.x * blockDim.x + threadIdx.x;
  long stride = (long)gridDim.x * blockDim.x;
  for (; i < total; i += stride) {
    long layer = i / matsz;
    long l = i - layer * matsz;
    int e = (int)(l & 15);
    int lane = (int)((l >> 4) & 31);
    long rest = l >> 9;
    int c  = (int)(rest % K32);
    int nt = (int)(rest / K32);
    int hx = lane & 15, g = lane >> 4;
    int vv = e >> 1, lo = e & 1;
    int k = c * 32 + kpair(vv, g) + lo;
    int n = nt * 16 + hx;
    out[i] = (__bf16)W[layer * matsz + (long)k * N + n];
  }
}

// ------- pack projected V into attn-B fragment order -------
// dest flat: ((((b*8+h)*32 + chunk)*2 + f)*32 + lane)*16 + e
//   s = chunk*32 + kpair(vv,g) + lo ; d = f*16 + hx
__global__ void pack_v(const __bf16* __restrict__ vb, __bf16* __restrict__ vp) {
  long total = (long)MROWS * DMOD;
  long i = (long)blockIdx.x * blockDim.x + threadIdx.x;
  long stride = (long)gridDim.x * blockDim.x;
  for (; i < total; i += stride) {
    int e    = (int)(i & 15);
    int lane = (int)((i >> 4) & 31);
    int f    = (int)((i >> 9) & 1);
    long rest = i >> 10;
    int chunk = (int)(rest & 31);
    int bh    = (int)(rest >> 5);
    int b = bh >> 3, h = bh & 7;
    int hx = lane & 15, g = lane >> 4;
    int vv = e >> 1, lo = e & 1;
    int s = chunk * 32 + kpair(vv, g) + lo;
    int d = f * 16 + hx;
    vp[i] = vb[((size_t)(b * SEQ + s)) * DMOD + h * DKH + d];
  }
}

// ---------------- WMMA GEMM:  C = A(MxK) @ W(KxN) + bias ----------------
// W pre-packed in fragment order. 4 waves/block; each wave owns a 16x64 strip
// (4 accumulators reuse one A fragment -> 4x less A traffic).
__global__ void gemm_bias(const __bf16* __restrict__ A, const __bf16* __restrict__ Wp,
                          const float* __restrict__ bias,
                          __bf16* __restrict__ outb, float* __restrict__ outf,
                          int M, int N, int K, int relu) {
  int lane = threadIdx.x & 31;
  int wave = threadIdx.x >> 5;
  int m0 = (blockIdx.y * 4 + wave) * 16;
  int nt0 = blockIdx.x * 4;
  int hx = lane & 15, g = lane >> 4;
  int K32 = K >> 5;

  const __bf16* ap = A + (size_t)(m0 + hx) * K;
  size_t tilestride = (size_t)K32 * 512;              // elems per packed n-tile
  const __bf16* wbase = Wp + (size_t)nt0 * tilestride + lane * 16;

  v8f acc[4];
#pragma unroll
  for (int t = 0; t < 4; ++t) acc[t] = (v8f){0.f,0.f,0.f,0.f,0.f,0.f,0.f,0.f};

  for (int c = 0; c < K32; ++c) {
    if (c + 2 < K32) __builtin_prefetch(ap + (c + 2) * 32, 0, 1);   // global_prefetch_b8
    v16bf a = load_a_frag(ap, c * 32, g);
#pragma unroll
    for (int t = 0; t < 4; ++t) {
      v16bf b = *(const v16bf*)(wbase + (size_t)t * tilestride + (size_t)c * 512);
      acc[t] = wmma_bf16(a, b, acc[t]);
    }
  }

#pragma unroll
  for (int t = 0; t < 4; ++t) {
    int n = (nt0 + t) * 16 + hx;
    float bb = bias[n];
#pragma unroll
    for (int vv = 0; vv < 8; ++vv) {
      int row = m0 + vv + 8 * g;
      float val = acc[t][vv] + bb;
      if (relu) val = fmaxf(val, 0.f);
      size_t idx = (size_t)row * N + n;
      if (outb) outb[idx] = (__bf16)val;
      if (outf) outf[idx] = val;
    }
  }
}

// ---------------- fused attention with distance decay ----------------
#define SPAD 1028                 // f32 score row stride (bank-conflict pad)
#define APAD 1032                 // bf16 attn row stride (16B-aligned for b128 LDS loads)
#define SEG  512
#define ATT_LDS (16 * SPAD * 4 + 16 * APAD * 2 + 64)

template <bool INCLUSIVE, bool ZERO_PAD>
__global__ void attn_kernel(const __bf16* __restrict__ Q,   // also K (kq_same)
                            const __bf16* __restrict__ Vp,  // fragment-packed V
                            __bf16* __restrict__ O,
                            const float* __restrict__ gammas, int layer) {
  extern __shared__ char smem[];
  float*  sc = (float*)smem;                                    // [16][SPAD]
  __bf16* at = (__bf16*)(smem + 16 * SPAD * 4);                 // [16][APAD]
  float*  dscale = (float*)(smem + 16 * SPAD * 4 + 16 * APAD * 2); // [16]

  int bh = blockIdx.y;
  int b = bh >> 3, h = bh & 7;
  int r0 = blockIdx.x * 16;
  int lane = threadIdx.x;
  int hx = lane & 15, g = lane >> 4;

  float gamma = gammas[layer * NHEAD + h];
  float gneg = -log1pf(__expf(gamma));            // -softplus(gamma)
  const float scale = 0.17677669529663687f;       // 1/sqrt(32)

  // ---- Q fragment ----
  const __bf16* qp = Q + ((size_t)(b * SEQ + r0 + hx)) * DMOD + h * DKH;
  v16bf aq = load_a_frag(qp, 0, g);

  // ---- scores strip: 64 WMMAs into LDS ----
  for (int j0 = 0; j0 < SEQ; j0 += 16) {
    const __bf16* kp = Q + ((size_t)(b * SEQ + j0 + hx)) * DMOD + h * DKH;
    if (j0 + 16 < SEQ) __builtin_prefetch(kp + 16 * DMOD, 0, 1);
    v16bf bk = load_a_frag(kp, 0, g);    // B fragment == A pattern (col = lane&15)
    v8f c = (v8f){0.f,0.f,0.f,0.f,0.f,0.f,0.f,0.f};
    c = wmma_bf16(aq, bk, c);
#pragma unroll
    for (int vv = 0; vv < 8; ++vv)
      sc[(vv + 8 * g) * SPAD + j0 + hx] = c[vv] * scale;
  }
  __syncthreads();

  // ---- softmax -> cumsum -> decay -> softmax2 (2 lanes per row) ----
  {
    int row = lane & 15;
    int hseg = lane >> 4;                   // 0: cols [0,512), 1: [512,1024)
    int i = r0 + row;
    float* srow = sc + row * SPAD;
    int lim = INCLUSIVE ? (i + 1) : i;
    if (lim > SEQ) lim = SEQ;
    int jbeg = hseg * SEG;
    int jend = jbeg + SEG; if (jend > lim) jend = lim;

    // walk 1: segment max
    float lmax = -3.0e38f;
    for (int j = jbeg; j < jend; ++j) lmax = fmaxf(lmax, srow[j]);
    float mx = fmaxf(lmax, __shfl_xor(lmax, 16, 32));

    // walk 2: segment exp-sum (at global max)
    float lsum = 0.f;
    for (int j = jbeg; j < jend; ++j) lsum += __expf(srow[j] - mx);
    float psum = __shfl_xor(lsum, 16, 32);  // partner segment sum
    float den = lsum + psum;
    float inv = (lim > 0) ? 1.f / den : 0.f;
    float tot = (lim > 0) ? 1.f : 0.f;

    // walk 3: cumsum + decay; cum offset for upper segment = lower-seg p mass
    float cum = hseg ? psum * inv : 0.f;
    float mx2 = -3.0e38f;
    for (int j = jbeg; j < jbeg + SEG; ++j) {
      float s = srow[j];
      bool valid = j < lim;
      float p = valid ? __expf(s - mx) * inv : 0.f;
      cum += p;
      float rem = fmaxf(tot - cum, 0.f);
      float d = sqrtf(rem * fabsf((float)(i - j)));
      float te = __expf(d * gneg);
      te = fminf(fmaxf(te, 1e-5f), 1e5f);
      float s2 = s * te;
      srow[j] = s2;
      if (valid) mx2 = fmaxf(mx2, s2);
    }
    mx2 = fmaxf(mx2, __shfl_xor(mx2, 16, 32));

    // walk 4: write UNNORMALIZED exp(s2-mx2) as bf16; normalization deferred to AV
    bool zerorow = (ZERO_PAD && i == 0) || (lim == 0);
    float lden2 = 0.f;
    __bf16* arow = at + row * APAD;
    for (int j = jbeg; j < jbeg + SEG; ++j) {
      float e2 = (!zerorow && j < lim) ? __expf(srow[j] - mx2) : 0.f;
      lden2 += e2;
      arow[j] = (__bf16)e2;
    }
    float den2 = lden2 + __shfl_xor(lden2, 16, 32);
    if (hseg == 0) dscale[row] = zerorow ? 0.f : 1.f / den2;
  }
  __syncthreads();

  // ---- out = attn(16xS) @ V(SxDK): A from LDS (b128 loads), B from packed V ----
  v8f c0 = (v8f){0.f,0.f,0.f,0.f,0.f,0.f,0.f,0.f};
  v8f c1 = (v8f){0.f,0.f,0.f,0.f,0.f,0.f,0.f,0.f};
  const __bf16* arow = at + hx * APAD;
  const __bf16* vpb = Vp + (((size_t)bh * 32) * 2 * 32 + lane) * 16;
  for (int ch = 0; ch < 32; ++ch) {               // 32 chunks of K=32
    v16bf aa = load_a_frag(arow, ch * 32, g);     // ds_load_b128 x2
    const __bf16* vc = vpb + (size_t)ch * 1024;   // chunk stride = 2*32*16
    v16bf b0 = *(const v16bf*)vc;                 // f=0 tile
    v16bf b1 = *(const v16bf*)(vc + 512);         // f=1 tile
    c0 = wmma_bf16(aa, b0, c0);
    c1 = wmma_bf16(aa, b1, c1);
  }
#pragma unroll
  for (int vv = 0; vv < 8; ++vv) {
    int rloc = vv + 8 * g;
    float s2 = dscale[rloc];                      // deferred softmax2 normalization
    int srw = r0 + rloc;
    size_t base = ((size_t)(b * SEQ + srw)) * DMOD + h * DKH;
    O[base + hx]      = (__bf16)(c0[vv] * s2);
    O[base + 16 + hx] = (__bf16)(c1[vv] * s2);
  }
}

// ---------------- fused residual-add + LayerNorm (one wave per row) ----------------
__global__ void add_ln(const float* __restrict__ x, const float* __restrict__ r,
                       const float* __restrict__ gw, const float* __restrict__ bw,
                       float* __restrict__ outf, __bf16* __restrict__ outb) {
  int row = blockIdx.x;
  int lane = threadIdx.x;
  const float* xr = x + (size_t)row * DMOD;
  const float* rr = r + (size_t)row * DMOD;

  float vals[8];
  float s = 0.f;
#pragma unroll
  for (int e = 0; e < 8; ++e) {
    int c = e * 32 + lane;
    float t = xr[c] + rr[c];
    vals[e] = t;
    s += t;
  }
#pragma unroll
  for (int m = 16; m >= 1; m >>= 1) s += __shfl_xor(s, m, 32);
  float mu = s * (1.f / 256.f);

  float v2 = 0.f;
#pragma unroll
  for (int e = 0; e < 8; ++e) { float d = vals[e] - mu; v2 += d * d; }
#pragma unroll
  for (int m = 16; m >= 1; m >>= 1) v2 += __shfl_xor(v2, m, 32);
  float rinv = rsqrtf(v2 * (1.f / 256.f) + 1e-5f);

#pragma unroll
  for (int e = 0; e < 8; ++e) {
    int c = e * 32 + lane;
    float o = (vals[e] - mu) * rinv * gw[c] + bw[c];
    outf[(size_t)row * DMOD + c] = o;
    outb[(size_t)row * DMOD + c] = (__bf16)o;
  }
}

// ---------------- host orchestration ----------------
struct BlockParams {
  const float *bk, *bv, *bo, *gammas, *ln1g, *ln1b, *b1, *b2, *ln2g, *ln2b;
  const __bf16 *wkb, *wvb, *wob, *w1b, *w2b;
  __bf16 *qb, *vb, *vp, *ob, *hb;
  float *pf;
};

static void run_block(hipStream_t stream, int layer,
                      float* xqf, __bf16* xqb, const __bf16* xvb,
                      bool inclusive, bool apply_pos,
                      float* final_f32, const BlockParams& P) {
  dim3 gP(DMOD / 64, MROWS / 64);   // N=256 GEMMs
  dim3 gF1(DFFN / 64, MROWS / 64);  // N=1024 GEMM
  dim3 blk(128);

  // q (== k, kq_same) and v projections -> bf16
  gemm_bias<<<gP, blk, 0, stream>>>(xqb, P.wkb + (size_t)layer * DMOD * DMOD,
                                    P.bk + layer * DMOD, P.qb, nullptr,
                                    MROWS, DMOD, DMOD, 0);
  gemm_bias<<<gP, blk, 0, stream>>>(xvb, P.wvb + (size_t)layer * DMOD * DMOD,
                                    P.bv + layer * DMOD, P.vb, nullptr,
                                    MROWS, DMOD, DMOD, 0);
  // repack V into WMMA fragment order
  pack_v<<<2048, 256, 0, stream>>>(P.vb, P.vp);

  // fused attention
  dim3 gA(SEQ / 16, BSZ * NHEAD);
  if (inclusive)
    attn_kernel<true, false><<<gA, 32, ATT_LDS, stream>>>(P.qb, P.vp, P.ob, P.gammas, layer);
  else
    attn_kernel<false, true><<<gA, 32, ATT_LDS, stream>>>(P.qb, P.vp, P.ob, P.gammas, layer);

  // output projection -> f32
  gemm_bias<<<gP, blk, 0, stream>>>(P.ob, P.wob + (size_t)layer * DMOD * DMOD,
                                    P.bo + layer * DMOD, nullptr, P.pf,
                                    MROWS, DMOD, DMOD, 0);

  // residual + LN1
  float* ln1_out = apply_pos ? xqf : final_f32;
  add_ln<<<MROWS, 32, 0, stream>>>(xqf, P.pf, P.ln1g + layer * DMOD,
                                   P.ln1b + layer * DMOD, ln1_out, xqb);

  if (apply_pos) {
    gemm_bias<<<gF1, blk, 0, stream>>>(xqb, P.w1b + (size_t)layer * DMOD * DFFN,
                                       P.b1 + layer * DFFN, P.hb, nullptr,
                                       MROWS, DFFN, DMOD, 1 /*relu*/);
    gemm_bias<<<gP, blk, 0, stream>>>(P.hb, P.w2b + (size_t)layer * DFFN * DMOD,
                                      P.b2 + layer * DMOD, nullptr, P.pf,
                                      MROWS, DMOD, DFFN, 0);
    add_ln<<<MROWS, 32, 0, stream>>>(xqf, P.pf, P.ln2g + layer * DMOD,
                                     P.ln2b + layer * DMOD, final_f32, xqb);
  }
}

extern "C" void kernel_launch(void* const* d_in, const int* in_sizes, int n_in,
                              void* d_out, int out_size, void* d_ws, size_t ws_size,
                              hipStream_t stream) {
  (void)in_sizes; (void)n_in; (void)out_size; (void)ws_size;

  const float* q_embed  = (const float*)d_in[0];
  const float* qa_embed = (const float*)d_in[1];
  const float* Wk = (const float*)d_in[2];
  const float* bk = (const float*)d_in[3];
  const float* Wv = (const float*)d_in[4];
  const float* bv = (const float*)d_in[5];
  const float* Wo = (const float*)d_in[6];
  const float* bo = (const float*)d_in[7];
  const float* gammas = (const float*)d_in[8];
  const float* ln1g = (const float*)d_in[9];
  const float* ln1b = (const float*)d_in[10];
  const float* W1 = (const float*)d_in[11];
  const float* b1 = (const float*)d_in[12];
  const float* W2 = (const float*)d_in[13];
  const float* b2 = (const float*)d_in[14];
  const float* ln2g = (const float*)d_in[15];
  const float* ln2b = (const float*)d_in[16];

  char* ws = (char*)d_ws;
  size_t off = 0;
  auto alloc = [&](size_t bytes) -> void* {
    void* p = ws + off;
    off = (off + bytes + 255) & ~(size_t)255;
    return p;
  };
  const size_t SDD = (size_t)NLAY * DMOD * DMOD;
  const size_t SDF = (size_t)NLAY * DMOD * DFFN;
  __bf16* wkb = (__bf16*)alloc(SDD * 2);
  __bf16* wvb = (__bf16*)alloc(SDD * 2);
  __bf16* wob = (__bf16*)alloc(SDD * 2);
  __bf16* w1b = (__bf16*)alloc(SDF * 2);
  __bf16* w2b = (__bf16*)alloc(SDF * 2);
  float*  xf  = (float*)alloc((size_t)MROWS * DMOD * 4);
  __bf16* xb  = (__bf16*)alloc((size_t)MROWS * DMOD * 2);
  float*  yf  = (float*)alloc((size_t)MROWS * DMOD * 4);
  __bf16* yb  = (__bf16*)alloc((size_t)MROWS * DMOD * 2);
  __bf16* qb  = (__bf16*)alloc((size_t)MROWS * DMOD * 2);
  __bf16* vb  = (__bf16*)alloc((size_t)MROWS * DMOD * 2);
  __bf16* vp  = (__bf16*)alloc((size_t)MROWS * DMOD * 2);
  __bf16* ob  = (__bf16*)alloc((size_t)MROWS * DMOD * 2);
  float*  pf  = (float*)alloc((size_t)MROWS * DMOD * 4);
  __bf16* hb  = (__bf16*)alloc((size_t)MROWS * DFFN * 2);

  // running activations (never mutate inputs)
  hipMemcpyAsync(xf, q_embed,  (size_t)MROWS * DMOD * 4, hipMemcpyDeviceToDevice, stream);
  hipMemcpyAsync(yf, qa_embed, (size_t)MROWS * DMOD * 4, hipMemcpyDeviceToDevice, stream);
  cvt_bf16<<<2048, 256, 0, stream>>>(q_embed,  xb, (long)MROWS * DMOD);
  cvt_bf16<<<2048, 256, 0, stream>>>(qa_embed, yb, (long)MROWS * DMOD);

  // weights -> bf16, packed into WMMA B-fragment order (recomputed each call)
  cvt_pack_w<<<2048, 256, 0, stream>>>(Wk, wkb, DMOD, DMOD, (long)SDD);
  cvt_pack_w<<<2048, 256, 0, stream>>>(Wv, wvb, DMOD, DMOD, (long)SDD);
  cvt_pack_w<<<2048, 256, 0, stream>>>(Wo, wob, DMOD, DMOD, (long)SDD);
  cvt_pack_w<<<2048, 256, 0, stream>>>(W1, w1b, DMOD, DFFN, (long)SDF);
  cvt_pack_w<<<2048, 256, 0, stream>>>(W2, w2b, DFFN, DMOD, (long)SDF);

  BlockParams P{bk, bv, bo, gammas, ln1g, ln1b, b1, b2, ln2g, ln2b,
                wkb, wvb, wob, w1b, w2b, qb, vb, vp, ob, hb, pf};

  // block 0: y = block(y,y,y) inclusive mask, FFN
  run_block(stream, 0, yf, yb, yb, true,  true,  yf, P);
  // block 1: x = block(x,x,x) inclusive mask, no FFN
  run_block(stream, 1, xf, xb, xb, true,  false, xf, P);
  // block 2: x = block(x,x,y) exclusive mask + zero_pad, FFN -> d_out
  run_block(stream, 2, xf, xb, yb, false, true,  (float*)d_out, P);
}
